// LSTMModel_1700807049229
// MI455X (gfx1250) — matrix operand order repaired and checked by reading, not compiled, not run on hardware
//
#include <hip/hip_runtime.h>
#include <hip/hip_bf16.h>

typedef __attribute__((ext_vector_type(16))) _Float16 v16h;
typedef __attribute__((ext_vector_type(8)))  _Float16 half8;
typedef __attribute__((ext_vector_type(8)))  float    v8f;
typedef __attribute__((ext_vector_type(4)))  float    f4;

#define T_STEPS 256
#define IN_DIM  24
#define HID     128
#define G4      512   // 4*HID

// ---- workspace byte offsets (f16 weights + combined f32 biases) ----
#define OFF_IH0 0                       // 512*32  f16 = 32768 B
#define OFF_HH0 32768                   // 512*128 f16 = 131072 B
#define OFF_IH1 163840
#define OFF_HH1 294912
#define OFF_IH2 425984
#define OFF_HH2 557056
#define OFF_B0  688128                  // 512 f32
#define OFF_B1  690176
#define OFF_B2  692224

// ---------------- prep kernels: f32 -> f16 weight conversion ----------------
__global__ void cvt_w_kernel(_Float16* __restrict__ dst, const float* __restrict__ src,
                             int kin, int kpad, int total) {
    int idx = blockIdx.x * blockDim.x + threadIdx.x;
    if (idx >= total) return;
    int n = idx / kpad, k = idx % kpad;
    dst[idx] = (k < kin) ? (_Float16)src[n * kin + k] : (_Float16)0.0f;
}

__global__ void add_bias_kernel(float* __restrict__ dst, const float* __restrict__ a,
                                const float* __restrict__ b, int n) {
    int idx = blockIdx.x * blockDim.x + threadIdx.x;
    if (idx < n) dst[idx] = a[idx] + b[idx];
}

// ---------------- device helpers ----------------
__device__ __forceinline__ v16h cat16(half8 lo, half8 hi) {
    return __builtin_shufflevector(lo, hi, 0,1,2,3,4,5,6,7,8,9,10,11,12,13,14,15);
}

// B operand: two 8-half runs per lane at p and p+16 (row-major W[N][Kpad])
__device__ __forceinline__ v16h load_b(const _Float16* p) {
    half8 lo = *(const half8*)p;
    half8 hi = *(const half8*)(p + 16);
    return cat16(lo, hi);
}

// A operand from LDS h tile [16][128] row-major: two contiguous 8-half runs
__device__ __forceinline__ v16h load_a(const _Float16* h, int kt, int m, int hi) {
    int base = m * HID + kt * 32 + hi * 8;
    half8 lo = *(const half8*)(h + base);
    half8 hp = *(const half8*)(h + base + 16);
    return cat16(lo, hp);
}

__device__ __forceinline__ v8f wmma(v16h a, v16h b, v8f c) {
    return __builtin_amdgcn_wmma_f32_16x16x32_f16(false, a, false, b, (short)0, c, false, false);
}

// 4 k-tiles x 4 gates, A from LDS, B from global (L2-resident f16 weights).
// kt loop deliberately NOT unrolled: limits live B tiles to 4 (32 VGPRs),
// avoiding the >256-VGPR + scratch-spill blow-up seen with full unroll.
__device__ __forceinline__ void mma4(v8f acc[4], const _Float16* a_lds,
                                     const _Float16* wrow, int m, int hi) {
#pragma unroll 1
    for (int kt = 0; kt < 4; ++kt) {
        v16h a = load_a(a_lds, kt, m, hi);
        const _Float16* wp = wrow + 32 * kt;
#pragma unroll
        for (int g = 0; g < 4; ++g)
            acc[g] = wmma(a, load_b(wp + (size_t)g * HID * HID), acc[g]);
    }
}

// fast sigmoid/tanh: single v_exp_f32 + single v_rcp_f32 (no IEEE div sequence)
__device__ __forceinline__ float sigm(float x) {
    return __builtin_amdgcn_rcpf(1.0f + __expf(-x));
}
__device__ __forceinline__ float tanhx(float x) {
#if __has_builtin(__builtin_amdgcn_tanhf)
    return __builtin_amdgcn_tanhf(x);        // gfx1250 hardware TANH trans op
#elif __has_builtin(__builtin_amdgcn_tanh_f32)
    return __builtin_amdgcn_tanh_f32(x);
#else
    return 2.0f * __builtin_amdgcn_rcpf(1.0f + __expf(-2.0f * x)) - 1.0f;
#endif
}

__device__ __forceinline__ void cell_update(v8f acc[4], v8f& c, _Float16* hdst,
                                            int j, int hi) {
#pragma unroll
    for (int r = 0; r < 8; ++r) {
        float iv = sigm(acc[0][r]);
        float fv = sigm(acc[1][r]);
        float gv = tanhx(acc[2][r]);
        float ov = sigm(acc[3][r]);
        float cv = fv * c[r] + iv * gv;
        c[r] = cv;
        float hv = ov * tanhx(cv);
        hdst[(r + hi * 8) * HID + j] = (_Float16)hv;
    }
}

__device__ __forceinline__ void init_acc(v8f acc[4], const float bias[4]) {
#pragma unroll
    for (int g = 0; g < 4; ++g)
#pragma unroll
        for (int r = 0; r < 8; ++r) acc[g][r] = bias[g];
}

// ---------------- fused 3-layer LSTM + FC ----------------
__global__ __launch_bounds__(256)
void lstm_fused_kernel(const float* __restrict__ x, const char* __restrict__ ws,
                       const float* __restrict__ Wfc, const float* __restrict__ bfc,
                       float* __restrict__ out) {
    const _Float16* Wih0 = (const _Float16*)(ws + OFF_IH0);
    const _Float16* Whh0 = (const _Float16*)(ws + OFF_HH0);
    const _Float16* Wih1 = (const _Float16*)(ws + OFF_IH1);
    const _Float16* Whh1 = (const _Float16*)(ws + OFF_HH1);
    const _Float16* Wih2 = (const _Float16*)(ws + OFF_IH2);
    const _Float16* Whh2 = (const _Float16*)(ws + OFF_HH2);
    const float* bias0 = (const float*)(ws + OFF_B0);
    const float* bias1 = (const float*)(ws + OFF_B1);
    const float* bias2 = (const float*)(ws + OFF_B2);

    // double-buffered h for the 3 layers: 3*2*16*128 f16 = 24 KB LDS
    __shared__ _Float16 hbuf[3][2][16 * HID];

    const int wave = threadIdx.x >> 5;
    const int lane = threadIdx.x & 31;
    const int m    = lane & 15;        // M row within tile
    const int hi   = lane >> 4;        // lane half (A/B layout split)
    const int j    = wave * 16 + m;    // hidden column owned by this (wave,lane)
    const int b0   = blockIdx.x * 16;  // batch tile base

    // zero initial h state
    for (int i = threadIdx.x; i < 3 * 2 * 16 * HID; i += 256)
        ((_Float16*)hbuf)[i] = (_Float16)0.0f;
    __syncthreads();

    // per-(wave,lane) biases, gate order i,f,g,o
    float b0v[4], b1v[4], b2v[4];
#pragma unroll
    for (int g = 0; g < 4; ++g) {
        b0v[g] = bias0[g * HID + j];
        b1v[g] = bias1[g * HID + j];
        b2v[g] = bias2[g * HID + j];
    }

    // per-lane weight row pointers: N = g*128 + j, B runs start at hi*8 and hi*8+16 within K
    const _Float16* pih0 = Wih0 + (size_t)j * 32  + hi * 8;  // Kpad=32 (layer 0 input)
    const _Float16* phh0 = Whh0 + (size_t)j * HID + hi * 8;
    const _Float16* pih1 = Wih1 + (size_t)j * HID + hi * 8;
    const _Float16* phh1 = Whh1 + (size_t)j * HID + hi * 8;
    const _Float16* pih2 = Wih2 + (size_t)j * HID + hi * 8;
    const _Float16* phh2 = Whh2 + (size_t)j * HID + hi * 8;

    v8f c0 = {0,0,0,0,0,0,0,0}, c1 = {0,0,0,0,0,0,0,0}, c2 = {0,0,0,0,0,0,0,0};

    const float* xrow = x + (size_t)(b0 + m) * T_STEPS * IN_DIM;
    // hi-lane mask for the padded K=24..31 region of layer-0 A tiles
    const float msk = (hi == 0) ? 1.0f : 0.0f;

    for (int t = 0; t < T_STEPS; ++t) {
        const int cur = t & 1, pv = cur ^ 1;
        v8f acc[4];

        // ---------------- layer 0 ----------------
        init_acc(acc, b0v);
        {
            // A tile from x (f32 -> f16), K padded 24 -> 32. Branchless:
            // run1 = K[hi*8 .. hi*8+8)  (always in-bounds)
            // run2 = K[16..24) for hi==0; zeros for hi==1 (loaded from a safe
            //        aliased in-row address, then masked to 0)
            const float* bx = xrow + t * IN_DIM;
            f4 r1a = *(const f4*)(bx + hi * 8);
            f4 r1b = *(const f4*)(bx + hi * 8 + 4);
            const float* bx2 = bx + (hi ? 8 : 16);   // in-bounds for both halves
            f4 r2a = *(const f4*)(bx2);
            f4 r2b = *(const f4*)(bx2 + 4);
            v16h a0;
#pragma unroll
            for (int i = 0; i < 4; ++i) {
                a0[i]      = (_Float16)r1a[i];
                a0[4 + i]  = (_Float16)r1b[i];
                a0[8 + i]  = (_Float16)(r2a[i] * msk);
                a0[12 + i] = (_Float16)(r2b[i] * msk);
            }
#pragma unroll
            for (int g = 0; g < 4; ++g)
                acc[g] = wmma(a0, load_b(pih0 + (size_t)g * HID * 32), acc[g]);
        }
        mma4(acc, &hbuf[0][pv][0], phh0, m, hi);
        cell_update(acc, c0, &hbuf[0][cur][0], j, hi);
        __syncthreads();

        // ---------------- layer 1 ----------------
        init_acc(acc, b1v);
        mma4(acc, &hbuf[0][cur][0], pih1, m, hi);   // input proj (h0 at t)
        mma4(acc, &hbuf[1][pv][0],  phh1, m, hi);   // hidden proj (h1 at t-1)
        cell_update(acc, c1, &hbuf[1][cur][0], j, hi);
        __syncthreads();

        // ---------------- layer 2 ----------------
        init_acc(acc, b2v);
        mma4(acc, &hbuf[1][cur][0], pih2, m, hi);
        mma4(acc, &hbuf[2][pv][0],  phh2, m, hi);
        cell_update(acc, c2, &hbuf[2][cur][0], j, hi);
        __syncthreads();
    }

    // ---------------- FC on final h2 ----------------
    if (threadIdx.x < 16) {
        const _Float16* hf = &hbuf[2][(T_STEPS - 1) & 1][0];
        float s = bfc[0];
#pragma unroll 8
        for (int k = 0; k < HID; ++k)
            s += (float)hf[threadIdx.x * HID + k] * Wfc[k];
        out[b0 + threadIdx.x] = s;
    }
}

// ---------------- host launch ----------------
extern "C" void kernel_launch(void* const* d_in, const int* in_sizes, int n_in,
                              void* d_out, int out_size, void* d_ws, size_t ws_size,
                              hipStream_t stream) {
    const float* x    = (const float*)d_in[0];
    const float* Wih0 = (const float*)d_in[1];
    const float* Whh0 = (const float*)d_in[2];
    const float* bih0 = (const float*)d_in[3];
    const float* bhh0 = (const float*)d_in[4];
    const float* Wih1 = (const float*)d_in[5];
    const float* Whh1 = (const float*)d_in[6];
    const float* bih1 = (const float*)d_in[7];
    const float* bhh1 = (const float*)d_in[8];
    const float* Wih2 = (const float*)d_in[9];
    const float* Whh2 = (const float*)d_in[10];
    const float* bih2 = (const float*)d_in[11];
    const float* bhh2 = (const float*)d_in[12];
    const float* Wfc  = (const float*)d_in[13];
    const float* bfc  = (const float*)d_in[14];
    float* out = (float*)d_out;
    char* ws = (char*)d_ws;

    const int B = in_sizes[0] / (T_STEPS * IN_DIM);   // 2048

    // --- prep: f32 -> f16 weights (L2-resident), combined biases ---
    {
        int n;
        n = G4 * 32;  cvt_w_kernel<<<(n + 255) / 256, 256, 0, stream>>>(
            (_Float16*)(ws + OFF_IH0), Wih0, IN_DIM, 32, n);
        n = G4 * HID; cvt_w_kernel<<<(n + 255) / 256, 256, 0, stream>>>(
            (_Float16*)(ws + OFF_HH0), Whh0, HID, HID, n);
        cvt_w_kernel<<<(n + 255) / 256, 256, 0, stream>>>(
            (_Float16*)(ws + OFF_IH1), Wih1, HID, HID, n);
        cvt_w_kernel<<<(n + 255) / 256, 256, 0, stream>>>(
            (_Float16*)(ws + OFF_HH1), Whh1, HID, HID, n);
        cvt_w_kernel<<<(n + 255) / 256, 256, 0, stream>>>(
            (_Float16*)(ws + OFF_IH2), Wih2, HID, HID, n);
        cvt_w_kernel<<<(n + 255) / 256, 256, 0, stream>>>(
            (_Float16*)(ws + OFF_HH2), Whh2, HID, HID, n);
        add_bias_kernel<<<2, 256, 0, stream>>>((float*)(ws + OFF_B0), bih0, bhh0, G4);
        add_bias_kernel<<<2, 256, 0, stream>>>((float*)(ws + OFF_B1), bih1, bhh1, G4);
        add_bias_kernel<<<2, 256, 0, stream>>>((float*)(ws + OFF_B2), bih2, bhh2, G4);
    }

    // --- fused recurrence: one WG per 16 batch rows, 8 waves (wave32) ---
    lstm_fused_kernel<<<B / 16, 256, 0, stream>>>(x, ws, Wfc, bfc, out);
}
